// FlashGPT2Attention_43490838839956
// MI455X (gfx1250) — matrix operand order repaired and checked by reading, not compiled, not run on hardware
//
#include <hip/hip_runtime.h>
#include <hip/hip_bf16.h>

// ---------------------------------------------------------------------------
// GPT-2 attention layer on MI455X (gfx1250), bf16 WMMA pipeline:
//   1) fp32 -> bf16 conversion (whole working set is L2-resident at 192MB)
//   2) QKV GEMM  : [8192,2048]x[2048,6144] + bias -> bf16 qkv   (v_wmma bf16)
//   3) Flash attn: causal, online softmax, K/V staged in LDS    (v_wmma bf16)
//   4) Proj GEMM : [8192,2048]x[2048,2048] + bias -> fp32 out   (v_wmma bf16)
// CDNA5 specifics: async global->LDS staging (ASYNCcnt), double-buffered GEMM,
// bank-conflict-free padded LDS layouts, transposed B/V tiles so every WMMA
// fragment gather is a pair of contiguous ds_load_b128.
// ---------------------------------------------------------------------------

typedef __attribute__((ext_vector_type(16))) __bf16 v16bf;
typedef __attribute__((ext_vector_type(8)))  float  v8f;
typedef int v4i32 __attribute__((vector_size(16)));   // b128 payload type

#define HIDDEN   2048
#define NHEADS   16
#define HEADDIM  128
#define SEQLEN   2048

// ---- CDNA5 async global->LDS copy (guarded; falls back to sync path) ------
#if defined(__has_builtin)
#if __has_builtin(__builtin_amdgcn_global_load_async_to_lds_b128)
#define HAVE_ASYNC_LDS 1
#endif
#if __has_builtin(__builtin_amdgcn_s_wait_asynccnt)
#define HAVE_WAIT_ASYNC 1
#endif
#endif

typedef __attribute__((address_space(1))) v4i32* gptr_b128;   // global src
typedef __attribute__((address_space(3))) v4i32* lptr_b128;   // LDS dst

__device__ __forceinline__ void copy16B_to_lds(const __bf16* g, __bf16* l) {
#ifdef HAVE_ASYNC_LDS
    __builtin_amdgcn_global_load_async_to_lds_b128((gptr_b128)g, (lptr_b128)l, 0, 0);
#else
    *(float4*)l = *(const float4*)g;
#endif
}

__device__ __forceinline__ void wait_async_all() {
#ifdef HAVE_ASYNC_LDS
#ifdef HAVE_WAIT_ASYNC
    __builtin_amdgcn_s_wait_asynccnt(0);
#else
    asm volatile("s_wait_asynccnt 0x0" ::: "memory");
#endif
#endif
}

// ---------------------------------------------------------------------------
// fp32 -> bf16 elementwise conversion
// ---------------------------------------------------------------------------
__global__ __launch_bounds__(256)
void f32_to_bf16_kernel(const float* __restrict__ in, __bf16* __restrict__ out, size_t n) {
    size_t i      = (size_t)blockIdx.x * blockDim.x + threadIdx.x;
    size_t stride = (size_t)gridDim.x * blockDim.x;
    for (; i < n; i += stride) out[i] = (__bf16)in[i];
}

// ---------------------------------------------------------------------------
// Tiled GEMM + bias: C[M,N] = A_bf16[M,K] * B_bf16[K,N] + bias[N]
// 256 threads = 8 waves; block tile 128x128; K-step 32; double-buffered LDS.
// Wave tile 32x64: 2 A-frags x 4 B-frags -> 8 v_wmma_f32_16x16x32_bf16/K-step.
// A staged row-major (async), B staged transposed [col][k] so B-fragment
// reads are contiguous. Rows padded to 40 elems (80B = 20 banks) so the 16
// lanes of a fragment land on disjoint bank quads.
// ---------------------------------------------------------------------------
template <typename OutT>
__global__ __launch_bounds__(256)
void gemm_bias_kernel(const __bf16* __restrict__ A, const __bf16* __restrict__ B,
                      const float* __restrict__ bias, OutT* __restrict__ C,
                      int M, int N, int K)
{
    __shared__ __align__(16) __bf16 As[2][128][40];   // row-major, padded  (20 KB)
    __shared__ __align__(16) __bf16 Bt[2][128][40];   // [col][k], padded   (20 KB)

    const int t    = threadIdx.x;
    const int wave = t >> 5;
    const int lane = t & 31;
    const int hf   = lane >> 4;
    const int lr   = lane & 15;

    const int m0 = blockIdx.x * 128;
    const int n0 = blockIdx.y * 128;
    const int wr = (wave >> 1) * 32;   // wave row offset: 0,32,64,96
    const int wc = (wave & 1) * 64;    // wave col offset: 0,64

    const v8f zero8 = {0.f,0.f,0.f,0.f,0.f,0.f,0.f,0.f};
    v8f acc[2][4];
#pragma unroll
    for (int a = 0; a < 2; ++a)
#pragma unroll
        for (int c = 0; c < 4; ++c) acc[a][c] = zero8;

    // A tile: 128x32 = 4096 elems, 16/thread (2 async 16B copies)
    const int ar = t >> 1;
    const int ac = (t & 1) * 16;
    // B tile: 32x128 = 4096 elems, 16/thread, transposing scatter
    const int bk = t >> 3;             // k row 0..31
    const int bc = (t & 7) * 16;       // col segment 0..112

    // ---- prologue: stage k0 = 0 into buffer 0
    {
        const __bf16* srcA = A + (size_t)(m0 + ar) * K + ac;
        copy16B_to_lds(srcA,     &As[0][ar][ac]);
        copy16B_to_lds(srcA + 8, &As[0][ar][ac + 8]);
        const __bf16* srcB = B + (size_t)bk * N + n0 + bc;
        float4 d0 = *(const float4*)(srcB);
        float4 d1 = *(const float4*)(srcB + 8);
        const __bf16* e0 = (const __bf16*)&d0;
        const __bf16* e1 = (const __bf16*)&d1;
#pragma unroll
        for (int i = 0; i < 8; ++i) {
            Bt[0][bc + i][bk]     = e0[i];
            Bt[0][bc + 8 + i][bk] = e1[i];
        }
    }
    wait_async_all();
    __syncthreads();

    int buf = 0;
    for (int k0 = 0; k0 < K; k0 += 32) {
        const bool more = (k0 + 32) < K;
        float4 d0, d1;
        // ---- issue next tile's global traffic first (A async, B to regs)
        if (more) {
            const __bf16* srcA = A + (size_t)(m0 + ar) * K + (k0 + 32) + ac;
            copy16B_to_lds(srcA,     &As[buf ^ 1][ar][ac]);
            copy16B_to_lds(srcA + 8, &As[buf ^ 1][ar][ac + 8]);
            const __bf16* srcB = B + (size_t)(k0 + 32 + bk) * N + n0 + bc;
            d0 = *(const float4*)(srcB);
            d1 = *(const float4*)(srcB + 8);
        }

        // ---- compute on current buffer
        v16bf af[2];
#pragma unroll
        for (int a = 0; a < 2; ++a) {
            int row = wr + a * 16 + lr;
#pragma unroll
            for (int j = 0; j < 8; ++j) {
                int kk = hf * 16 + 2 * j;
                af[a][2*j]     = As[buf][row][kk];
                af[a][2*j + 1] = As[buf][row][kk + 1];
            }
        }
#pragma unroll
        for (int c = 0; c < 4; ++c) {
            v16bf bfv;
            int col = wc + c * 16 + lr;
#pragma unroll
            for (int j = 0; j < 8; ++j) {
                bfv[2*j]     = Bt[buf][col][hf * 16 + 2 * j];
                bfv[2*j + 1] = Bt[buf][col][hf * 16 + 2 * j + 1];
            }
            acc[0][c] = __builtin_amdgcn_wmma_f32_16x16x32_bf16(
                false, af[0], false, bfv, (short)0, acc[0][c], false, false);
            acc[1][c] = __builtin_amdgcn_wmma_f32_16x16x32_bf16(
                false, af[1], false, bfv, (short)0, acc[1][c], false, false);
        }

        // ---- finish staging next B tile (transposing scatter)
        if (more) {
            const __bf16* e0 = (const __bf16*)&d0;
            const __bf16* e1 = (const __bf16*)&d1;
#pragma unroll
            for (int i = 0; i < 8; ++i) {
                Bt[buf ^ 1][bc + i][bk]     = e0[i];
                Bt[buf ^ 1][bc + 8 + i][bk] = e1[i];
            }
        }
        wait_async_all();
        __syncthreads();
        buf ^= 1;
    }

    // ---- epilogue: bias + store (C layout: M = r + 8*hf, N = lane%16)
#pragma unroll
    for (int a = 0; a < 2; ++a) {
#pragma unroll
        for (int c = 0; c < 4; ++c) {
            int col  = n0 + wc + c * 16 + lr;
            float bv = bias[col];
#pragma unroll
            for (int r = 0; r < 8; ++r) {
                int row = m0 + wr + a * 16 + r + hf * 8;
                C[(size_t)row * N + col] = (OutT)(acc[a][c][r] + bv);
            }
        }
    }
}

// ---------------------------------------------------------------------------
// Causal flash attention, bf16 WMMA, online softmax.
// Grid: (seq/64, NHEADS, B). Block: 128 threads = 4 waves, 16 q-rows/wave.
// K staged row-major+padded (async); V staged transposed [d][key] so P*V
// B-fragments are contiguous ds_load_b128.
// ---------------------------------------------------------------------------
__global__ __launch_bounds__(128)
void flash_attn_kernel(const __bf16* __restrict__ qkv, __bf16* __restrict__ attn_out,
                       int seq_len)
{
    __shared__ __align__(16) __bf16 Ks[32][136];     // [key][d], padded (8.5 KB)
    __shared__ __align__(16) __bf16 Vt[128][40];     // [d][key], padded (10 KB)
    __shared__ __align__(16) __bf16 Pb[4][16][40];   // per-wave P staging (5 KB)

    const int t    = threadIdx.x;
    const int wave = t >> 5;
    const int lane = t & 31;
    const int hf   = lane >> 4;
    const int lr   = lane & 15;

    const int b   = blockIdx.z;
    const int h   = blockIdx.y;
    const int q0  = blockIdx.x * 64;
    const int qr0 = q0 + wave * 16;

    const size_t rowstride = 3 * HIDDEN;                 // 6144
    const size_t qoff = (size_t)h * HEADDIM;
    const size_t koff = HIDDEN + (size_t)h * HEADDIM;
    const size_t voff = 2 * HIDDEN + (size_t)h * HEADDIM;
    const float  scale = 0.08838834764831845f;           // 1/sqrt(128)

    // ---- Q fragments: 4 x (16x32) covering head_dim 128 (contiguous b128s)
    v16bf qf[4];
    {
        size_t base = (size_t)(b * seq_len + qr0 + lr) * rowstride + qoff;
#pragma unroll
        for (int c = 0; c < 4; ++c) {
#pragma unroll
            for (int j = 0; j < 8; ++j) {
                int kk = c * 32 + hf * 16 + 2 * j;
                qf[c][2*j]     = qkv[base + kk];
                qf[c][2*j + 1] = qkv[base + kk + 1];
            }
        }
    }

    const v8f zero8 = {0.f,0.f,0.f,0.f,0.f,0.f,0.f,0.f};
    v8f o[8];
#pragma unroll
    for (int c = 0; c < 8; ++c) o[c] = zero8;
    float m[8], l[8];
#pragma unroll
    for (int r = 0; r < 8; ++r) { m[r] = -1e30f; l[r] = 0.f; }

    const int ntiles = blockIdx.x * 2 + 2;   // 32-key tiles covering keys <= q0+63
    for (int kt = 0; kt < ntiles; ++kt) {
        const int kb = kt * 32;

        // ---- stage K (async, row-major) and V (transposing scatter)
#pragma unroll
        for (int s2 = 0; s2 < 2; ++s2) {
            int sid = t * 2 + s2;
            int row = sid >> 3;            // key 0..31
            int cs  = (sid & 7) * 16;      // d segment 0..112
            size_t g = (size_t)(b * seq_len + kb + row) * rowstride;
            copy16B_to_lds(qkv + g + koff + cs,     &Ks[row][cs]);
            copy16B_to_lds(qkv + g + koff + cs + 8, &Ks[row][cs + 8]);
            float4 d0 = *(const float4*)(qkv + g + voff + cs);
            float4 d1 = *(const float4*)(qkv + g + voff + cs + 8);
            const __bf16* e0 = (const __bf16*)&d0;
            const __bf16* e1 = (const __bf16*)&d1;
#pragma unroll
            for (int i = 0; i < 8; ++i) {
                Vt[cs + i][row]     = e0[i];
                Vt[cs + 8 + i][row] = e1[i];
            }
        }
        wait_async_all();
        __syncthreads();

        // ---- scores S = Q * K^T for 32 keys (two 16-key chunks)
        v8f s[2];
#pragma unroll
        for (int ks2 = 0; ks2 < 2; ++ks2) {
            v8f sc = zero8;
#pragma unroll
            for (int c = 0; c < 4; ++c) {
                v16bf kf;                              // B: (d x key) 32x16
#pragma unroll
                for (int j = 0; j < 8; ++j) {
                    int dk = c * 32 + hf * 16 + 2 * j;
                    kf[2*j]     = Ks[ks2 * 16 + lr][dk];
                    kf[2*j + 1] = Ks[ks2 * 16 + lr][dk + 1];
                }
                sc = __builtin_amdgcn_wmma_f32_16x16x32_bf16(
                    false, qf[c], false, kf, (short)0, sc, false, false);
            }
            s[ks2] = sc;
        }

        // ---- scale + causal mask
#pragma unroll
        for (int ks2 = 0; ks2 < 2; ++ks2) {
            int key = kb + ks2 * 16 + lr;
#pragma unroll
            for (int r = 0; r < 8; ++r) {
                int row = qr0 + r + hf * 8;
                float v = s[ks2][r] * scale;
                s[ks2][r] = (key > row) ? -1e30f : v;
            }
        }

        // ---- online softmax (row lives in a 16-lane half; xor 1..8 stays inside)
#pragma unroll
        for (int r = 0; r < 8; ++r) {
            float mt = fmaxf(s[0][r], s[1][r]);
            mt = fmaxf(mt, __shfl_xor(mt, 1, 32));
            mt = fmaxf(mt, __shfl_xor(mt, 2, 32));
            mt = fmaxf(mt, __shfl_xor(mt, 4, 32));
            mt = fmaxf(mt, __shfl_xor(mt, 8, 32));
            float mn = fmaxf(m[r], mt);
            float f  = __expf(m[r] - mn);
            m[r] = mn;
            float p0 = __expf(s[0][r] - mn);
            float p1 = __expf(s[1][r] - mn);
            float rs = p0 + p1;
            rs += __shfl_xor(rs, 1, 32);
            rs += __shfl_xor(rs, 2, 32);
            rs += __shfl_xor(rs, 4, 32);
            rs += __shfl_xor(rs, 8, 32);
            l[r] = l[r] * f + rs;
            s[0][r] = p0;
            s[1][r] = p1;
#pragma unroll
            for (int c = 0; c < 8; ++c) o[c][r] *= f;
        }

        // ---- P (C-layout) -> per-wave LDS -> A-fragment (same-wave LDS in-order)
#pragma unroll
        for (int r = 0; r < 8; ++r) {
            Pb[wave][r + hf * 8][lr]      = (__bf16)s[0][r];
            Pb[wave][r + hf * 8][16 + lr] = (__bf16)s[1][r];
        }
        v16bf pa;
#pragma unroll
        for (int j = 0; j < 8; ++j) {
            int kk = hf * 16 + 2 * j;
            pa[2*j]     = Pb[wave][lr][kk];
            pa[2*j + 1] = Pb[wave][lr][kk + 1];
        }

        // ---- O += P * V   (8 d-chunks; Vt gives contiguous B-fragments)
#pragma unroll
        for (int c = 0; c < 8; ++c) {
            v16bf vf;                                  // B: (key x d) 32x16
#pragma unroll
            for (int j = 0; j < 8; ++j) {
                vf[2*j]     = Vt[c * 16 + lr][hf * 16 + 2 * j];
                vf[2*j + 1] = Vt[c * 16 + lr][hf * 16 + 2 * j + 1];
            }
            o[c] = __builtin_amdgcn_wmma_f32_16x16x32_bf16(
                false, pa, false, vf, (short)0, o[c], false, false);
        }
        __syncthreads();
    }

    // ---- normalize and store bf16 attention output [T, HIDDEN]
#pragma unroll
    for (int c = 0; c < 8; ++c) {
#pragma unroll
        for (int r = 0; r < 8; ++r) {
            int trow = b * seq_len + qr0 + r + hf * 8;
            attn_out[(size_t)trow * HIDDEN + h * HEADDIM + c * 16 + lr] =
                (__bf16)(o[c][r] / l[r]);
        }
    }
}

// ---------------------------------------------------------------------------
// Host-side launcher
// ---------------------------------------------------------------------------
extern "C" void kernel_launch(void* const* d_in, const int* in_sizes, int n_in,
                              void* d_out, int out_size, void* d_ws, size_t ws_size,
                              hipStream_t stream) {
    const float* hs     = (const float*)d_in[0];   // [T, 2048]
    const float* w_attn = (const float*)d_in[1];   // [2048, 6144]
    const float* b_attn = (const float*)d_in[2];   // [6144]
    const float* w_proj = (const float*)d_in[3];   // [2048, 2048]
    const float* b_proj = (const float*)d_in[4];   // [2048]
    float* out = (float*)d_out;

    const int T   = in_sizes[0] / HIDDEN;          // 8192
    const int seq = SEQLEN;                        // 2048 (static in reference setup)
    const int B   = T / seq;                       // 4

    // workspace layout (bf16 buffers)
    char* ws = (char*)d_ws;
    __bf16* hs_bf    = (__bf16*)(ws);                                            // T*2048
    __bf16* wattn_bf = (__bf16*)(ws + (size_t)T * HIDDEN * 2);                   // 2048*6144
    __bf16* wproj_bf = (__bf16*)((char*)wattn_bf + (size_t)HIDDEN * 3 * HIDDEN * 2);
    __bf16* qkv_bf   = (__bf16*)((char*)wproj_bf + (size_t)HIDDEN * HIDDEN * 2); // T*6144
    __bf16* attn_bf  = (__bf16*)((char*)qkv_bf + (size_t)T * 3 * HIDDEN * 2);    // T*2048

    // 1) fp32 -> bf16 conversions
    f32_to_bf16_kernel<<<2048, 256, 0, stream>>>(hs, hs_bf, (size_t)T * HIDDEN);
    f32_to_bf16_kernel<<<2048, 256, 0, stream>>>(w_attn, wattn_bf, (size_t)HIDDEN * 3 * HIDDEN);
    f32_to_bf16_kernel<<<1024, 256, 0, stream>>>(w_proj, wproj_bf, (size_t)HIDDEN * HIDDEN);

    // 2) QKV GEMM -> bf16 qkv [T, 6144]
    {
        dim3 grid(T / 128, (3 * HIDDEN) / 128);
        gemm_bias_kernel<__bf16><<<grid, 256, 0, stream>>>(
            hs_bf, wattn_bf, b_attn, qkv_bf, T, 3 * HIDDEN, HIDDEN);
    }

    // 3) causal flash attention -> bf16 attn [T, 2048]
    {
        dim3 grid(seq / 64, NHEADS, B);
        flash_attn_kernel<<<grid, 128, 0, stream>>>(qkv_bf, attn_bf, seq);
    }

    // 4) output projection -> fp32 d_out [T, 2048]
    {
        dim3 grid(T / 128, HIDDEN / 128);
        gemm_bias_kernel<float><<<grid, 256, 0, stream>>>(
            attn_bf, wproj_bf, b_proj, out, T, HIDDEN, HIDDEN);
    }
}
